// MultichannelTransformerDecoder_7284264534714
// MI455X (gfx1250) — matrix operand order repaired and checked by reading, not compiled, not run on hardware
//
#include <hip/hip_runtime.h>
#include <hip/hip_bf16.h>

// ---------------------------------------------------------------------------
// MultichannelTransformerDecoder for MI455X (gfx1250, wave32, WMMA).
// One LDS-staged bf16 WMMA GEMM serves QK^T, P*V and klin/vlin:
//   - all GEMM operands are produced in bf16 with the k-axis contiguous, so
//     tile staging is per-lane 16B copies -> GLOBAL_LOAD_ASYNC_TO_LDS_B128
//     (ASYNCcnt + s_wait_asynccnt) when the builtin exists, plain
//     b128 load + ds_store_b128 otherwise
//   - 256 threads (8 waves), 64x128 macro-tile, BK=32; each wave computes
//     16x64 via 4 accumulators sharing one A fragment (4 wmma : 10 ds_load)
//   - fragments = 2x ds_load_b128 per operand (CDNA5 16x16x32 lane maps)
//   - v_wmma_f32_16x16x32_bf16, f32 accumulation
// ---------------------------------------------------------------------------

#define Cc   8
#define Fdim 1024
#define Wd   512
#define Hh   8
#define Dh   128      // Fdim / Hh
#define MCc  4
#define MFf  512
#define CFW  (Cc * Fdim * Wd)          // 4,194,304
#define MCFW (Cc * MFf * Wd)
#define QKN  (Cc * Hh * Wd * Wd)       // 16,777,216
#define LINN (Cc * Fdim * MFf)         // klin/vlin elements

typedef __attribute__((ext_vector_type(16))) __bf16 v16bf;
typedef __attribute__((ext_vector_type(8)))  __bf16 v8bf;
typedef __attribute__((ext_vector_type(8)))  float  v8f;
typedef __attribute__((ext_vector_type(4)))  int    v4i;

#if defined(__has_builtin)
#if __has_builtin(__builtin_amdgcn_global_load_async_to_lds_b128) && \
    __has_builtin(__builtin_amdgcn_s_wait_asynccnt)
#define USE_ASYNC_LDS 1
#endif
#endif

static __device__ __forceinline__ __bf16 f2bf(float f) {
  union { float f; unsigned u; } uf; uf.f = f;
  unsigned r = uf.u + 0x7FFFu + ((uf.u >> 16) & 1u);   // round-to-nearest-even
  union { unsigned short s; __bf16 b; } ub;
  ub.s = (unsigned short)(r >> 16);
  return ub.b;
}

static __device__ __forceinline__ v8f wmma_bf16(v16bf a, v16bf b, v8f c) {
  return __builtin_amdgcn_wmma_f32_16x16x32_bf16(
      false, a, false, b, (short)0, c, false, false);
}

// 16-byte global -> LDS tile copy (per lane). Async path uses the CDNA5
// async-load-to-LDS instruction tracked by ASYNCcnt.
static __device__ __forceinline__ void stage_cp(__bf16* dst, const __bf16* src) {
#ifdef USE_ASYNC_LDS
  __builtin_amdgcn_global_load_async_to_lds_b128(
      (v4i __attribute__((address_space(1)))*)(unsigned long long)(uintptr_t)src,
      (v4i __attribute__((address_space(3)))*)(unsigned)(uintptr_t)dst,
      0, 0);
#else
  *(v8bf*)dst = *(const v8bf*)src;
#endif
}

static __device__ __forceinline__ void stage_fence() {
#ifdef USE_ASYNC_LDS
  __builtin_amdgcn_s_wait_asynccnt(0);
#endif
}

// ---------------------------------------------------------------------------
// Batched GEMM: C[bz] = scale * A[bz] x B[bz]
//   A element (m,k): A + (bz/div)*bsA1 + (bz%div)*bsA2 + m*sAm + k   (bf16)
//   B element (k,n): B + (bz/div)*bsB1 + (bz%div)*bsB2 + n*sBn + k   (bf16)
//   C element (m,n): C + (bz/div)*bsC1 + (bz%div)*bsC2 + m*sCm + n*sCn
// M % 64 == 0, N % 128 == 0, K % 32 == 0. outBf16 selects fp32/bf16 C.
// ---------------------------------------------------------------------------
#define BM 64
#define BN 128
#define BK 32
#define LDP 40   // padded LDS row (bf16); 80B rows keep 16B-aligned segments

__global__ __launch_bounds__(256)
void gemm_wmma_kernel(const __bf16* __restrict__ A, int sAm, long long bsA1, long long bsA2,
                      const __bf16* __restrict__ B, int sBn, long long bsB1, long long bsB2,
                      void* __restrict__ Cout, int sCm, int sCn, long long bsC1, long long bsC2,
                      int Kdim, int bdiv, float scale, int outBf16) {
  __shared__ __align__(16) __bf16 As[BM][LDP];  // [m][k]
  __shared__ __align__(16) __bf16 Bs[BN][LDP];  // [n][k]

  const int tid = threadIdx.x;
  const int lane = tid & 31, wid = tid >> 5;
  const int mt = wid & 3;       // wave's 16-row m tile
  const int nq = wid >> 2;      // wave's 64-col n quad (0..1)
  const int lm = lane & 15, hi = lane >> 4;

  const int bz = blockIdx.z;
  const int bo = bz / bdiv, bi = bz % bdiv;
  const __bf16* Ab = A + (long long)bo * bsA1 + (long long)bi * bsA2
                       + (size_t)(blockIdx.x * BM) * sAm;
  const __bf16* Bb = B + (long long)bo * bsB1 + (long long)bi * bsB2
                       + (size_t)(blockIdx.y * BN) * sBn;

  const int ar = tid >> 2, asg = (tid & 3) * 8;   // A: 64 rows x 4 segs = 256

  v8f acc[4] = {};

  for (int k0 = 0; k0 < Kdim; k0 += BK) {
    stage_cp(&As[ar][asg], Ab + (size_t)ar * sAm + (k0 + asg));
#pragma unroll
    for (int i = 0; i < 2; ++i) {                 // B: 128 rows x 4 segs = 512
      int u = tid + 256 * i;
      int r = u >> 2, sg = (u & 3) * 8;
      stage_cp(&Bs[r][sg], Bb + (size_t)r * sBn + (k0 + sg));
    }
    stage_fence();
    __syncthreads();

    v16bf a;
    {
      const int am = mt * 16 + lm;
      v8bf lo = *(const v8bf*)&As[am][hi * 8];
      v8bf h8 = *(const v8bf*)&As[am][16 + hi * 8];
#pragma unroll
      for (int i = 0; i < 8; ++i) { a[i] = lo[i]; a[i + 8] = h8[i]; }
    }
#pragma unroll
    for (int j = 0; j < 4; ++j) {
      const int bn = nq * 64 + j * 16 + lm;
      v8bf lo = *(const v8bf*)&Bs[bn][hi * 8];
      v8bf h8 = *(const v8bf*)&Bs[bn][16 + hi * 8];
      v16bf b;
#pragma unroll
      for (int i = 0; i < 8; ++i) { b[i] = lo[i]; b[i + 8] = h8[i]; }
      acc[j] = wmma_bf16(a, b, acc[j]);
    }
    __syncthreads();
  }

  // C/D lane map: vgpr r -> row r+8*hi, col lane&15
  const long long cb = (long long)bo * bsC1 + (long long)bi * bsC2;
  const int mrow = blockIdx.x * BM + mt * 16 + 8 * hi;
  if (outBf16) {
    __bf16* Cb = (__bf16*)Cout;
#pragma unroll
    for (int j = 0; j < 4; ++j) {
      int ncol = blockIdx.y * BN + nq * 64 + j * 16 + lm;
#pragma unroll
      for (int r = 0; r < 8; ++r)
        Cb[cb + (long long)(mrow + r) * sCm + (long long)ncol * sCn] = f2bf(acc[j][r] * scale);
    }
  } else {
    float* Cb = (float*)Cout;
#pragma unroll
    for (int j = 0; j < 4; ++j) {
      int ncol = blockIdx.y * BN + nq * 64 + j * 16 + lm;
#pragma unroll
      for (int r = 0; r < 8; ++r)
        Cb[cb + (long long)(mrow + r) * sCm + (long long)ncol * sCn] = acc[j][r] * scale;
    }
  }
}

// ---------------------------------------------------------------------------
// LayerNorm over F (=1024) per (c,w), per-(c,f) affine. 256 threads.
__global__ void ln_kernel(const float* __restrict__ x, const float* __restrict__ g,
                          const float* __restrict__ b, float* __restrict__ y) {
  int c = blockIdx.x >> 9;
  int w = blockIdx.x & (Wd - 1);
  const float* xp = x + (size_t)c * Fdim * Wd + w;
  float* yp = y + (size_t)c * Fdim * Wd + w;
  __shared__ float red[256];
  int t = threadIdx.x;
  float vals[4], s = 0.f, s2 = 0.f;
#pragma unroll
  for (int i = 0; i < 4; ++i) {
    float v = xp[(size_t)(t + i * 256) * Wd];
    vals[i] = v; s += v; s2 += v * v;
  }
  red[t] = s; __syncthreads();
  for (int off = 128; off > 0; off >>= 1) { if (t < off) red[t] += red[t + off]; __syncthreads(); }
  float tot = red[0]; __syncthreads();
  red[t] = s2; __syncthreads();
  for (int off = 128; off > 0; off >>= 1) { if (t < off) red[t] += red[t + off]; __syncthreads(); }
  float tot2 = red[0];
  float mu = tot * (1.0f / Fdim);
  float var = tot2 * (1.0f / Fdim) - mu * mu;
  float rstd = rsqrtf(var + 1e-8f);
#pragma unroll
  for (int i = 0; i < 4; ++i) {
    int f = t + i * 256;
    yp[(size_t)f * Wd] = (vals[i] - mu) * rstd * g[c * Fdim + f] + b[c * Fdim + f];
  }
}

// 1 x ksz conv along W. Optional fp32 out (normal layout) and/or bf16 out
// (normal [oc][f][w] or transposed [oc][w][f] when ybTr != 0).
__global__ void conv1xk_kernel(const float* __restrict__ x, const float* __restrict__ wt,
                               const float* __restrict__ bias,
                               float* __restrict__ yf, __bf16* __restrict__ yb, int ybTr,
                               int IC, int OCn, int Fn, int ksz) {
  int idx = blockIdx.x * blockDim.x + threadIdx.x;
  int total = OCn * Fn * Wd;
  if (idx >= total) return;
  int w = idx & (Wd - 1);
  int f = (idx / Wd) % Fn;
  int oc = idx / (Wd * Fn);
  int pad = ksz >> 1;
  float acc = bias[oc];
  for (int ic = 0; ic < IC; ++ic) {
    const float* xp = x + ((size_t)ic * Fn + f) * Wd;
    const float* wp = wt + ((size_t)oc * IC + ic) * ksz;
    for (int tt = 0; tt < ksz; ++tt) {
      int ww = w + tt - pad;
      if (ww >= 0 && ww < Wd) acc += xp[ww] * wp[tt];
    }
  }
  if (yf) yf[idx] = acc;
  if (yb) {
    if (ybTr) yb[((size_t)oc * Wd + w) * Fn + f] = f2bf(acc);
    else      yb[idx] = f2bf(acc);
  }
}

// 3x3 conv over (F,W), pad 1; act=1 -> squared ReLU
__global__ void conv3x3_kernel(const float* __restrict__ x, const float* __restrict__ wt,
                               const float* __restrict__ bias, float* __restrict__ y,
                               int IC, int OCn, int act) {
  int idx = blockIdx.x * blockDim.x + threadIdx.x;
  int total = OCn * Fdim * Wd;
  if (idx >= total) return;
  int w = idx & (Wd - 1);
  int f = (idx / Wd) & (Fdim - 1);
  int oc = idx / (Wd * Fdim);
  float acc = bias[oc];
  for (int ic = 0; ic < IC; ++ic) {
    const float* xp = x + (size_t)ic * Fdim * Wd;
    const float* wp = wt + ((size_t)oc * IC + ic) * 9;
#pragma unroll
    for (int df = -1; df <= 1; ++df) {
      int ff = f + df;
      if (ff < 0 || ff >= Fdim) continue;
#pragma unroll
      for (int dw = -1; dw <= 1; ++dw) {
        int ww = w + dw;
        if (ww < 0 || ww >= Wd) continue;
        acc += xp[(size_t)ff * Wd + ww] * wp[(df + 1) * 3 + (dw + 1)];
      }
    }
  }
  if (act) { acc = fmaxf(acc, 0.f); acc = acc * acc; }
  y[idx] = acc;
}

__global__ void add_inplace_kernel(float* __restrict__ a, const float* __restrict__ b, int n) {
  int i = blockIdx.x * blockDim.x + threadIdx.x;
  if (i < n) a[i] += b[i];
}

__global__ void cast_bf16_kernel(const float* __restrict__ src, __bf16* __restrict__ dst, int n) {
  int i = blockIdx.x * blockDim.x + threadIdx.x;
  if (i < n) dst[i] = f2bf(src[i]);
}

// softmax over last axis (Wd = 512); bf16 probabilities out
__global__ void softmax_kernel(const float* __restrict__ L, __bf16* __restrict__ P) {
  int row = blockIdx.x;
  const float* x = L + (size_t)row * Wd;
  __bf16* p = P + (size_t)row * Wd;
  __shared__ float red[256];
  int t = threadIdx.x;
  float v0 = x[t], v1 = x[t + 256];
  red[t] = fmaxf(v0, v1); __syncthreads();
  for (int off = 128; off > 0; off >>= 1) { if (t < off) red[t] = fmaxf(red[t], red[t + off]); __syncthreads(); }
  float m = red[0]; __syncthreads();
  float e0 = __expf(v0 - m), e1 = __expf(v1 - m);
  red[t] = e0 + e1; __syncthreads();
  for (int off = 128; off > 0; off >>= 1) { if (t < off) red[t] += red[t + off]; __syncthreads(); }
  float inv = 1.0f / red[0];
  p[t] = f2bf(e0 * inv);
  p[t + 256] = f2bf(e1 * inv);
}

// ---------------------------------------------------------------------------
// Host-side orchestration
// ---------------------------------------------------------------------------
static inline void launch_gemm(hipStream_t s,
                               const __bf16* A, int sAm, long long bsA1, long long bsA2,
                               const __bf16* B, int sBn, long long bsB1, long long bsB2,
                               void* C, int sCm, int sCn, long long bsC1, long long bsC2,
                               int M, int N, int K, int batch, int bdiv,
                               float scale, int outBf16) {
  gemm_wmma_kernel<<<dim3(M / BM, N / BN, batch), 256, 0, s>>>(
      A, sAm, bsA1, bsA2, B, sBn, bsB1, bsB2, C, sCm, sCn, bsC1, bsC2,
      K, bdiv, scale, outBf16);
}

static inline void launch_conv1xk(hipStream_t s, const float* x, const float* wt,
                                  const float* bias, float* yf, __bf16* yb, int ybTr,
                                  int IC, int OCn, int Fn, int ksz) {
  int total = OCn * Fn * Wd;
  conv1xk_kernel<<<(total + 255) / 256, 256, 0, s>>>(x, wt, bias, yf, yb, ybTr,
                                                     IC, OCn, Fn, ksz);
}

struct AttnParams {
  const float *qw, *qb, *kw, *kb, *vw, *vb, *ow, *ob, *klin, *vlin;
};

static void attention_block(hipStream_t s,
                            float* H, float* LN, float* Z,
                            __bf16* QT, __bf16* KTb, __bf16* VBb,
                            __bf16* KTc, __bf16* VTc, __bf16* LKB, __bf16* LVB,
                            __bf16* P,
                            const float* normw, const float* normb,
                            const AttnParams& p,
                            const float* src,   // nullptr => self (use LN)
                            int srcC, int srcF, int ksz,
                            float* logits) {
  ln_kernel<<<Cc * Wd, 256, 0, s>>>(H, normw, normb, LN);
  // Q transposed bf16: QT[c][w][F]
  launch_conv1xk(s, LN, p.qw, p.qb, nullptr, QT, 1, Cc, Cc, Fdim, 3);
  const float* S = src ? src : LN;
  if (p.klin) {
    // conv K/V -> transposed [c][w][MF] (klin B operand, k=m contiguous)
    launch_conv1xk(s, S, p.kw, p.kb, nullptr, KTc, 1, srcC, Cc, srcF, ksz);
    cast_bf16_kernel<<<(LINN + 255) / 256, 256, 0, s>>>(p.klin, LKB, LINN);
    // KTb[c][w][f] = sum_m klin[c][f][m] * KTc[c][w][m]   (bf16 out, transposed)
    launch_gemm(s, LKB, MFf, (long long)Fdim * MFf, 0,
                KTc, MFf, (long long)Wd * MFf, 0,
                KTb, 1, Fdim, (long long)Wd * Fdim, 0,
                Fdim, Wd, MFf, Cc, 1, 1.0f, 1);
    launch_conv1xk(s, S, p.vw, p.vb, nullptr, VTc, 1, srcC, Cc, srcF, ksz);
    cast_bf16_kernel<<<(LINN + 255) / 256, 256, 0, s>>>(p.vlin, LVB, LINN);
    // VBb[c][f][w] (normal layout, bf16)
    launch_gemm(s, LVB, MFf, (long long)Fdim * MFf, 0,
                VTc, MFf, (long long)Wd * MFf, 0,
                VBb, Wd, 1, (long long)Fdim * Wd, 0,
                Fdim, Wd, MFf, Cc, 1, 1.0f, 1);
  } else {
    launch_conv1xk(s, S, p.kw, p.kb, nullptr, KTb, 1, srcC, Cc, srcF, ksz);  // [c][w][F]
    launch_conv1xk(s, S, p.vw, p.vb, nullptr, VBb, 0, srcC, Cc, srcF, ksz);  // [c][f][w]
  }
  // logits[ch][wq][wk] = 1/32 * sum_d QT[c][wq][h*Dh+d] * KTb[c][wk][h*Dh+d]
  launch_gemm(s, QT, Fdim, (long long)Wd * Fdim, Dh,
              KTb, Fdim, (long long)Wd * Fdim, Dh,
              logits, Wd, 1, (long long)Hh * Wd * Wd, (long long)Wd * Wd,
              Wd, Wd, Dh, Cc * Hh, Hh, 0.03125f, 0);
  softmax_kernel<<<Cc * Hh * Wd, 256, 0, s>>>(logits, P);
  // attn_out[c][h*Dh+d][wq] = sum_wk P[ch][wq][wk] * VBb[c][h*Dh+d][wk]
  launch_gemm(s, P, Wd, (long long)Hh * Wd * Wd, (long long)Wd * Wd,
              VBb, Wd, (long long)Fdim * Wd, (long long)Dh * Wd,
              LN, 1, Wd, (long long)Fdim * Wd, (long long)Dh * Wd,
              Wd, Dh, Wd, Cc * Hh, Hh, 1.0f, 0);
  launch_conv1xk(s, LN, p.ow, p.ob, Z, nullptr, 0, Cc, Cc, Fdim, 3);
  add_inplace_kernel<<<CFW / 256, 256, 0, s>>>(H, Z, CFW);
}

extern "C" void kernel_launch(void* const* d_in, const int* in_sizes, int n_in,
                              void* d_out, int out_size, void* d_ws, size_t ws_size,
                              hipStream_t stream) {
  (void)in_sizes; (void)n_in; (void)out_size; (void)ws_size;
  // Flatten order assumed: top-level insertion order (x, skip, cross, cross2,
  // params), params tree-flattened with sorted keys at each dict level:
  //   attn{1..4}: kb,[klin],kw,ob,ow,qb,qw,vb,[vlin],vw
  //   conv1_b, conv1_w, conv2_b, conv2_w, norm{1,2,3,4,6}_{b,w}
  const float* x      = (const float*)d_in[0];
  const float* skip   = (const float*)d_in[1];
  const float* cross  = (const float*)d_in[2];
  const float* cross2 = (const float*)d_in[3];

  auto F = [&](int i) { return (const float*)d_in[i]; };
  AttnParams a1 = { F(9),  F(8),  F(5),  F(4),  F(11), F(10), F(7),  F(6),  nullptr, nullptr };
  AttnParams a2 = { F(17), F(16), F(13), F(12), F(19), F(18), F(15), F(14), nullptr, nullptr };
  AttnParams a3 = { F(26), F(25), F(22), F(20), F(29), F(27), F(24), F(23), F(21),  F(28)  };
  AttnParams a4 = { F(36), F(35), F(32), F(30), F(39), F(37), F(34), F(33), F(31),  F(38)  };
  const float* conv1_b = F(40); const float* conv1_w = F(41);
  const float* conv2_b = F(42); const float* conv2_w = F(43);
  const float* n1b = F(44); const float* n1w = F(45);
  const float* n2b = F(46); const float* n2w = F(47);
  const float* n3b = F(48); const float* n3w = F(49);
  const float* n4b = F(50); const float* n4w = F(51);
  const float* n6b = F(52); const float* n6w = F(53);

  // Workspace layout (float units; all bf16 buffers start 16B-aligned)
  float* ws  = (float*)d_ws;
  float* H   = ws;                         // [C,F,W] fp32
  float* LN  = H  + CFW;                   // [C,F,W] fp32 (also attn output)
  float* Z   = LN + CFW;                   // [C,F,W] fp32 (proj / FFN out)
  float* SC  = Z  + CFW;                   // [QKN] fp32: attn2 logits / FFN hidden
  __bf16* QT  = (__bf16*)(SC + QKN);       // [C,W,F] bf16
  __bf16* KTb = QT  + CFW;                 // [C,W,F] bf16
  __bf16* VBb = KTb + CFW;                 // [C,F,W] bf16
  __bf16* KTc = VBb + CFW;                 // [C,W,MF] bf16
  __bf16* VTc = KTc + MCFW;                // [C,W,MF] bf16
  __bf16* LKB = VTc + MCFW;                // [C,F,MF] bf16 (klin cast)
  __bf16* LVB = LKB + LINN;                // [C,F,MF] bf16 (vlin cast)
  __bf16* P   = LVB + LINN;                // [C,H,Wq,Wk] bf16 probabilities

  // d_out sections: h, qk1(attn1), qk2(attn3), qk3(attn4)
  float* out_h   = (float*)d_out;
  float* out_qk1 = out_h + CFW;
  float* out_qk2 = out_qk1 + QKN;
  float* out_qk3 = out_qk2 + QKN;

  (void)hipMemcpyAsync(H, x, (size_t)CFW * sizeof(float), hipMemcpyDeviceToDevice, stream);

  attention_block(stream, H, LN, Z, QT, KTb, VBb, KTc, VTc, LKB, LVB, P,
                  n1w, n1b, a1, /*src=*/nullptr, Cc, Fdim, 3, out_qk1);
  attention_block(stream, H, LN, Z, QT, KTb, VBb, KTc, VTc, LKB, LVB, P,
                  n2w, n2b, a2, skip, Cc, Fdim, 3, SC);
  attention_block(stream, H, LN, Z, QT, KTb, VBb, KTc, VTc, LKB, LVB, P,
                  n3w, n3b, a3, cross, MCc, MFf, 5, out_qk2);
  attention_block(stream, H, LN, Z, QT, KTb, VBb, KTc, VTc, LKB, LVB, P,
                  n4w, n4b, a4, cross2, MCc, MFf, 5, out_qk3);

  // FFN: LN -> conv3x3(8->32) + squared ReLU -> conv3x3(32->8) -> residual
  ln_kernel<<<Cc * Wd, 256, 0, stream>>>(H, n6w, n6b, LN);
  {
    int tot1 = (Cc * 4) * Fdim * Wd;
    conv3x3_kernel<<<(tot1 + 255) / 256, 256, 0, stream>>>(LN, conv1_w, conv1_b, SC, Cc, Cc * 4, 1);
    int tot2 = Cc * Fdim * Wd;
    conv3x3_kernel<<<(tot2 + 255) / 256, 256, 0, stream>>>(SC, conv2_w, conv2_b, Z, Cc * 4, Cc, 0);
  }
  add_inplace_kernel<<<CFW / 256, 256, 0, stream>>>(H, Z, CFW);

  (void)hipMemcpyAsync(out_h, H, (size_t)CFW * sizeof(float), hipMemcpyDeviceToDevice, stream);
}
